// TransformerBlockMoE_52338471469503
// MI455X (gfx1250) — matrix operand order repaired and checked by reading, not compiled, not run on hardware
//
#include <hip/hip_runtime.h>

typedef _Float16 half_t;
typedef __attribute__((ext_vector_type(16))) _Float16 v16h;
typedef __attribute__((ext_vector_type(8)))  float    v8f;

#define BATCH 2
#define SEQ   1024
#define HD    1024
#define NHEAD 16
#define DHEAD 64
#define NEXP  8
#define FDIM  4096
#define NTOK  (BATCH * SEQ)

union HFrag { v16h h; unsigned u[8]; uint4 q[2]; _Float16 e[16]; };

__device__ __forceinline__ v8f wmma16(const HFrag& a, const HFrag& b, v8f c) {
  // D = A(16x32 f16) * B(32x16 f16) + C(16x16 f32)
  return __builtin_amdgcn_wmma_f32_16x16x32_f16(false, a.h, false, b.h, (short)0, c,
                                                false, false);
}

// CDNA5 async global->LDS copy (ASYNCcnt-tracked), 16B per lane.
__device__ __forceinline__ void async_ld_b128(unsigned lds_byte_off, const void* gaddr) {
  asm volatile("global_load_async_to_lds_b128 %0, %1, off"
               :: "v"(lds_byte_off), "v"(gaddr) : "memory");
}
__device__ __forceinline__ void wait_async0() {
  asm volatile("s_wait_asynccnt 0" ::: "memory");
}
__device__ __forceinline__ unsigned lds_off32(const void* p) {
  return (unsigned)(unsigned long long)p;  // low 32 bits of generic = LDS offset
}

// ---------------------------------------------------------------------------
// Elementwise helpers
// ---------------------------------------------------------------------------
__global__ __launch_bounds__(256) void cvt_f16_kernel(const float* __restrict__ s,
                                                      half_t* __restrict__ d, int n) {
  int i = blockIdx.x * 256 + threadIdx.x;
  if (i < n) d[i] = (half_t)s[i];
}

__global__ __launch_bounds__(256) void residual_kernel(const float* __restrict__ x,
                                                       const float* __restrict__ y,
                                                       float* __restrict__ o, int n) {
  int i = blockIdx.x * 256 + threadIdx.x;
  if (i < n) o[i] = x[i] + y[i];
}

__global__ __launch_bounds__(256) void silu_mul_kernel(const float* __restrict__ g1,
                                                       const float* __restrict__ g3,
                                                       half_t* __restrict__ d, int n) {
  int i = blockIdx.x * 256 + threadIdx.x;
  if (i < n) {
    float a = g1[i];
    float s = a / (1.0f + __expf(-a));
    d[i] = (half_t)(s * g3[i]);
  }
}

__global__ __launch_bounds__(256) void moe_accum_kernel(float* __restrict__ o,
                                                        const float* __restrict__ ye,
                                                        const float* __restrict__ we,
                                                        int e, int n) {
  int i = blockIdx.x * 256 + threadIdx.x;
  if (i < n) {
    int row = i >> 10;  // / HD
    o[i] += we[row * NEXP + e] * ye[i];
  }
}

// [N, H] f32  ->  [B, NH, S, 64] f16
__global__ __launch_bounds__(256) void permute_qkv_kernel(const float* __restrict__ s,
                                                          half_t* __restrict__ d, int n) {
  int i = blockIdx.x * 256 + threadIdx.x;
  if (i < n) {
    int dd = i & 63;
    int ss = (i >> 6) & (SEQ - 1);
    int hh = (i >> 16) & (NHEAD - 1);
    int bb = i >> 20;
    d[i] = (half_t)s[(size_t)(bb * SEQ + ss) * HD + hh * DHEAD + dd];
  }
}

// ---------------------------------------------------------------------------
// LayerNorm: one block per row, fp32 in -> fp16 out
// ---------------------------------------------------------------------------
__global__ __launch_bounds__(256) void ln_kernel(const float* __restrict__ X,
                                                 const float* __restrict__ g,
                                                 const float* __restrict__ bta,
                                                 half_t* __restrict__ Y) {
  int row = blockIdx.x;
  const float* x = X + (size_t)row * HD;
  __shared__ float s1[256], s2[256];
  float a = 0.f, b2 = 0.f;
  for (int c = threadIdx.x; c < HD; c += 256) { float v = x[c]; a += v; b2 += v * v; }
  s1[threadIdx.x] = a; s2[threadIdx.x] = b2;
  __syncthreads();
  for (int off = 128; off > 0; off >>= 1) {
    if (threadIdx.x < off) {
      s1[threadIdx.x] += s1[threadIdx.x + off];
      s2[threadIdx.x] += s2[threadIdx.x + off];
    }
    __syncthreads();
  }
  float mu = s1[0] * (1.0f / HD);
  float var = s2[0] * (1.0f / HD) - mu * mu;
  float rs = rsqrtf(var + 1e-5f);
  for (int c = threadIdx.x; c < HD; c += 256)
    Y[(size_t)row * HD + c] = (half_t)((x[c] - mu) * rs * g[c] + bta[c]);
}

// ---------------------------------------------------------------------------
// WMMA GEMM:  C[M,N] f32 = A[M,K] f16 x W[N,K]^T f16   (torch Linear)
// Block: 128 thr = 4 waves (2x2), block tile 128x128, wave tile 64x64 (4x4 frags),
// K step 32.  Staging: global_load_async_to_lds_b128, fenced by s_wait_asynccnt.
// LDS rows padded to 20 dwords so fragment reads are 16B-aligned ds_load_b128.
// ---------------------------------------------------------------------------
#define LDT 20  // dwords per LDS row (32 halfs data + pad); 80B rows keep 16B align

__global__ __launch_bounds__(128) void gemm_f16_kernel(const half_t* __restrict__ A,
                                                       const half_t* __restrict__ W,
                                                       float* __restrict__ C,
                                                       int M, int N, int K) {
  __shared__ unsigned AsU[128 * LDT];
  __shared__ unsigned BsU[128 * LDT];
  const int tid = threadIdx.x, lane = tid & 31, wave = tid >> 5;
  const int m0 = blockIdx.y * 128, n0 = blockIdx.x * 128;
  const int wm = (wave >> 1) * 64, wn = (wave & 1) * 64;
  const int ln = lane & 15, kh = lane >> 4;

  const v8f zero = {0.f, 0.f, 0.f, 0.f, 0.f, 0.f, 0.f, 0.f};
  v8f acc[4][4];
  for (int i = 0; i < 4; ++i)
    for (int j = 0; j < 4; ++j) acc[i][j] = zero;

  // staging: each thread issues 4 x 16B async loads per tile
  const int r0 = tid >> 2;            // 0..31
  const int scol = (tid & 3) * 8;     // 0,8,16,24 halfs
  const unsigned ldsA = lds_off32(&AsU[0]);
  const unsigned ldsB = lds_off32(&BsU[0]);

  for (int kt = 0; kt < K; kt += 32) {
    __syncthreads();  // protect LDS reuse from previous iteration
#pragma unroll
    for (int i = 0; i < 4; ++i) {
      int row = r0 + 32 * i;
      async_ld_b128(ldsA + (unsigned)(row * LDT + (scol >> 1)) * 4u,
                    A + (size_t)(m0 + row) * K + kt + scol);
      async_ld_b128(ldsB + (unsigned)(row * LDT + (scol >> 1)) * 4u,
                    W + (size_t)(n0 + row) * K + kt + scol);
    }
    if (kt + 32 < K) {
      __builtin_prefetch(A + (size_t)(m0 + r0) * K + kt + 32 + scol, 0, 1);
      __builtin_prefetch(W + (size_t)(n0 + r0) * K + kt + 32 + scol, 0, 1);
    }
    wait_async0();
    __syncthreads();

    HFrag a[4];
#pragma unroll
    for (int i = 0; i < 4; ++i) {
      int rb = wm + i * 16 + ln;
      // A frag: u[v] = K-pair (v/4)*16 + kh*8 + (v%4)*2  -> two aligned b128 reads
      a[i].q[0] = *(const uint4*)&AsU[rb * LDT + kh * 4];
      a[i].q[1] = *(const uint4*)&AsU[rb * LDT + 8 + kh * 4];
    }
#pragma unroll
    for (int j = 0; j < 4; ++j) {
      int cb = wn + j * 16 + ln;
      HFrag b;
      // B frag: u[v] = K-pair (v/4)*16 + kh*8 + (v%4)*2 -> two aligned b128 reads
      b.q[0] = *(const uint4*)&BsU[cb * LDT + kh * 4];
      b.q[1] = *(const uint4*)&BsU[cb * LDT + 8 + kh * 4];
#pragma unroll
      for (int i = 0; i < 4; ++i)
        acc[i][j] = wmma16(a[i], b, acc[i][j]);
    }
  }

  for (int i = 0; i < 4; ++i)
    for (int j = 0; j < 4; ++j)
      for (int v = 0; v < 8; ++v) {
        int row = m0 + wm + i * 16 + v + 8 * kh;
        int col = n0 + wn + j * 16 + ln;
        C[(size_t)row * N + col] = acc[i][j][v];
      }
}

// ---------------------------------------------------------------------------
// Causal attention, one block per (b, head, 16-query tile), 4 waves.
// Q,K,V: [B,NH,S,64] f16.  O: [N,H] f16 (concat heads).
// LDS: scores f32[16][1024] | probs f16[16][1024] | reduce scratch
// ---------------------------------------------------------------------------
#define ATT_SC_BYTES (16 * SEQ * 4)
#define ATT_PF_BYTES (16 * SEQ * 2)
#define ATT_SMEM (ATT_SC_BYTES + ATT_PF_BYTES + 160 * 4)

__global__ __launch_bounds__(128) void attn_kernel(const half_t* __restrict__ Q,
                                                   const half_t* __restrict__ K,
                                                   const half_t* __restrict__ V,
                                                   half_t* __restrict__ O) {
  const int qtile = blockIdx.x;
  const int h = blockIdx.y, b = blockIdx.z;
  const int tid = threadIdx.x, lane = tid & 31, wave = tid >> 5;
  const int q0 = qtile * 16;

  extern __shared__ char smem[];
  float*  sc     = (float*)smem;                                  // [16][SEQ]
  half_t* pf     = (half_t*)(smem + ATT_SC_BYTES);                // [16][SEQ]
  float*  red    = (float*)(smem + ATT_SC_BYTES + ATT_PF_BYTES);  // [16][8]
  float*  rowmax = red + 128;
  float*  rowsum = rowmax + 16;

  const size_t bh = (size_t)b * NHEAD + h;
  const half_t* Qb = Q + (bh * SEQ + q0) * DHEAD;
  const half_t* Kb = K + bh * SEQ * DHEAD;
  const half_t* Vb = V + bh * SEQ * DHEAD;

  const int m = lane & 15, kh = lane >> 4;
  const v8f zero = {0.f, 0.f, 0.f, 0.f, 0.f, 0.f, 0.f, 0.f};

  // Q fragments: 16x64 as two 16x32 A-frags (same in every wave)
  HFrag qa[2];
  for (int c = 0; c < 2; ++c)
    for (int v = 0; v < 8; ++v) {
      int g = v >> 2, vv = v & 3;
      int k0 = c * 32 + g * 16 + kh * 8 + vv * 2;
      qa[c].u[v] = *(const unsigned*)&Qb[m * DHEAD + k0];
    }

  // Phase 1: scores = Q K^T / 8 with causal mask, tiles split across waves
  const float scale = 0.125f;  // 1/sqrt(64)
  for (int t = wave; t <= qtile; t += 4) {
    HFrag kb[2];
    for (int c = 0; c < 2; ++c)
      for (int v = 0; v < 8; ++v) {
        int k0 = c * 32 + (v >> 2) * 16 + kh * 8 + (v & 3) * 2;  // B frag: pairs of d for key m
        kb[c].u[v] = *(const unsigned*)&Kb[(size_t)(t * 16 + m) * DHEAD + k0];
      }
    v8f s = zero;
    s = wmma16(qa[0], kb[0], s);
    s = wmma16(qa[1], kb[1], s);
    for (int v = 0; v < 8; ++v) {
      int row = v + 8 * kh;  // local query row
      int col = t * 16 + m;  // absolute key
      float val = s[v] * scale;
      sc[row * SEQ + col] = (col <= q0 + row) ? val : -1e30f;
    }
  }
  __syncthreads();

  // Phase 2: exact softmax per row (8 threads per row), pack probs to f16
  {
    int r = tid >> 3, j = tid & 7;
    int nv = q0 + r + 1;
    float mx = -1e30f;
    for (int c = j; c < nv; c += 8) mx = fmaxf(mx, sc[r * SEQ + c]);
    red[r * 8 + j] = mx;
    __syncthreads();
    if (j == 0) {
      float m2 = red[r * 8];
      for (int q = 1; q < 8; ++q) m2 = fmaxf(m2, red[r * 8 + q]);
      rowmax[r] = m2;
    }
    __syncthreads();
    float rm = rowmax[r], sum = 0.f;
    for (int c = j; c < nv; c += 8) {
      float e = __expf(sc[r * SEQ + c] - rm);
      sc[r * SEQ + c] = e;
      sum += e;
    }
    red[r * 8 + j] = sum;
    __syncthreads();
    if (j == 0) {
      float s2 = 0.f;
      for (int q = 0; q < 8; ++q) s2 += red[r * 8 + q];
      rowsum[r] = s2;
    }
    __syncthreads();
    float inv = 1.f / rowsum[r];
    for (int c = j; c < SEQ; c += 8)
      pf[r * SEQ + c] = (half_t)((c < nv) ? sc[r * SEQ + c] * inv : 0.f);
  }
  __syncthreads();

  // Phase 3: O = P V, each wave owns one 16-wide d-slice, keys are K-dim
  int kchunks = (qtile + 2) >> 1;  // ceil((q0+16)/32)
  int nb = wave * 16;
  v8f o = zero;
  for (int ch = 0; ch < kchunks; ++ch) {
    HFrag pa, vb;
    for (int v = 0; v < 8; ++v) {
      int g = v >> 2, vv = v & 3;
      int k0 = ch * 32 + g * 16 + kh * 8 + vv * 2;
      pa.u[v] = *(const unsigned*)&pf[m * SEQ + k0];
      int kk = ch * 32 + g * 16 + kh * 8 + vv * 2;  // two consecutive keys, same d
      unsigned lo = *(const unsigned short*)&Vb[(size_t)kk * DHEAD + nb + m];
      unsigned hi = *(const unsigned short*)&Vb[(size_t)(kk + 1) * DHEAD + nb + m];
      vb.u[v] = lo | (hi << 16);
    }
    o = wmma16(pa, vb, o);
  }
  for (int v = 0; v < 8; ++v) {
    int row = v + 8 * kh, col = nb + m;
    O[((size_t)b * SEQ + q0 + row) * HD + h * DHEAD + col] = (half_t)o[v];
  }
}

// ---------------------------------------------------------------------------
// Router: logits [N,8], top-2 gate weights, load-balance loss
// ---------------------------------------------------------------------------
__global__ __launch_bounds__(256) void router_logits_kernel(const half_t* __restrict__ h2,
                                                            const float* __restrict__ Wr,
                                                            float* __restrict__ logits) {
  int wave = threadIdx.x >> 5, lane = threadIdx.x & 31;
  int row = blockIdx.x * 8 + wave;
  float acc[NEXP];
  for (int e = 0; e < NEXP; ++e) acc[e] = 0.f;
  for (int c = lane; c < HD; c += 32) {
    float xv = (float)h2[(size_t)row * HD + c];
    for (int e = 0; e < NEXP; ++e) acc[e] += xv * Wr[e * HD + c];
  }
  for (int e = 0; e < NEXP; ++e)
    for (int off = 16; off > 0; off >>= 1)
      acc[e] += __shfl_xor(acc[e], off, 32);
  if (lane == 0)
    for (int e = 0; e < NEXP; ++e) logits[(size_t)row * NEXP + e] = acc[e];
}

__global__ __launch_bounds__(256) void routing_kernel(const float* __restrict__ logits,
                                                      float* __restrict__ we) {
  int n = blockIdx.x * 256 + threadIdx.x;
  if (n >= NTOK) return;
  float l[NEXP];
  for (int e = 0; e < NEXP; ++e) l[e] = logits[n * NEXP + e];
  int i0 = 0;
  for (int e = 1; e < NEXP; ++e) if (l[e] > l[i0]) i0 = e;
  int i1 = (i0 == 0) ? 1 : 0;
  for (int e = 0; e < NEXP; ++e) if (e != i0 && l[e] > l[i1]) i1 = e;
  float e1 = __expf(l[i1] - l[i0]);
  float w0 = 1.f / (1.f + e1);
  float w1 = 1.f - w0;
  for (int e = 0; e < NEXP; ++e)
    we[n * NEXP + e] = (e == i0) ? w0 : ((e == i1) ? w1 : 0.f);
}

__global__ __launch_bounds__(256) void lb_kernel(const float* __restrict__ logits,
                                                 float* __restrict__ out_lb) {
  __shared__ float su[NEXP][256];
  float u[NEXP];
  for (int e = 0; e < NEXP; ++e) u[e] = 0.f;
  for (int n = threadIdx.x; n < NTOK; n += 256) {
    float l[NEXP]; float mx = -1e30f;
    for (int e = 0; e < NEXP; ++e) { l[e] = logits[n * NEXP + e]; mx = fmaxf(mx, l[e]); }
    float s = 0.f;
    for (int e = 0; e < NEXP; ++e) { l[e] = __expf(l[e] - mx); s += l[e]; }
    float inv = 1.f / s;
    for (int e = 0; e < NEXP; ++e) u[e] += l[e] * inv;
  }
  for (int e = 0; e < NEXP; ++e) su[e][threadIdx.x] = u[e];
  __syncthreads();
  for (int off = 128; off > 0; off >>= 1) {
    if (threadIdx.x < off)
      for (int e = 0; e < NEXP; ++e) su[e][threadIdx.x] += su[e][threadIdx.x + off];
    __syncthreads();
  }
  if (threadIdx.x == 0) {
    float acc = 0.f;
    for (int e = 0; e < NEXP; ++e) { float ue = su[e][0] / (float)NTOK; acc += ue * ue; }
    out_lb[0] = (float)NEXP * acc;
  }
}

// ---------------------------------------------------------------------------
// Host orchestration
// ---------------------------------------------------------------------------
extern "C" void kernel_launch(void* const* d_in, const int* in_sizes, int n_in,
                              void* d_out, int out_size, void* d_ws, size_t ws_size,
                              hipStream_t stream) {
  const float* x     = (const float*)d_in[0];
  const float* Wq    = (const float*)d_in[1];
  const float* Wk    = (const float*)d_in[2];
  const float* Wv    = (const float*)d_in[3];
  const float* Wo    = (const float*)d_in[4];
  const float* ln1_g = (const float*)d_in[5];
  const float* ln1_b = (const float*)d_in[6];
  const float* ln2_g = (const float*)d_in[7];
  const float* ln2_b = (const float*)d_in[8];
  const float* Wr    = (const float*)d_in[9];
  const float* W1    = (const float*)d_in[10];
  const float* W2    = (const float*)d_in[11];
  const float* W3    = (const float*)d_in[12];
  float* out = (float*)d_out;

  char* p = (char*)d_ws;
  auto carve = [&](size_t bytes) -> void* {
    void* r = (void*)p;
    p += (bytes + 255) & ~((size_t)255);
    return r;
  };

  const size_t NH_ = (size_t)NTOK * HD;    // 2M
  const size_t NF_ = (size_t)NTOK * FDIM;  // 8M
  const size_t WHH = (size_t)HD * HD;      // 1M
  const size_t WFH = (size_t)FDIM * HD;    // 4M

  half_t* Wq_h   = (half_t*)carve(WHH * 2);
  half_t* Wk_h   = (half_t*)carve(WHH * 2);
  half_t* Wv_h   = (half_t*)carve(WHH * 2);
  half_t* Wo_h   = (half_t*)carve(WHH * 2);
  half_t* h1_h   = (half_t*)carve(NH_ * 2);
  float*  qf     = (float*) carve(NH_ * 4);
  float*  kf     = (float*) carve(NH_ * 4);
  float*  vf     = (float*) carve(NH_ * 4);
  half_t* qh     = (half_t*)carve(NH_ * 2);
  half_t* kh     = (half_t*)carve(NH_ * 2);
  half_t* vh     = (half_t*)carve(NH_ * 2);
  half_t* attn_h = (half_t*)carve(NH_ * 2);
  float*  proj   = (float*) carve(NH_ * 4);
  half_t* h2_h   = (half_t*)carve(NH_ * 2);
  float*  logits = (float*) carve((size_t)NTOK * NEXP * 4);
  float*  we     = (float*) carve((size_t)NTOK * NEXP * 4);
  half_t* w1e_h  = (half_t*)carve(WFH * 2);
  half_t* w3e_h  = (half_t*)carve(WFH * 2);
  half_t* w2e_h  = (half_t*)carve(WFH * 2);
  float*  g1     = (float*) carve(NF_ * 4);
  float*  g3     = (float*) carve(NF_ * 4);
  half_t* hg_h   = (half_t*)carve(NF_ * 2);
  float*  ye     = (float*) carve(NH_ * 4);

  const int EW = 256;
  auto blocks = [](size_t n) { return (int)((n + 255) / 256); };

  // 1) attention weights -> f16
  cvt_f16_kernel<<<blocks(WHH), EW, 0, stream>>>(Wq, Wq_h, (int)WHH);
  cvt_f16_kernel<<<blocks(WHH), EW, 0, stream>>>(Wk, Wk_h, (int)WHH);
  cvt_f16_kernel<<<blocks(WHH), EW, 0, stream>>>(Wv, Wv_h, (int)WHH);
  cvt_f16_kernel<<<blocks(WHH), EW, 0, stream>>>(Wo, Wo_h, (int)WHH);

  // 2) LN1
  ln_kernel<<<NTOK, EW, 0, stream>>>(x, ln1_g, ln1_b, h1_h);

  // 3) Q/K/V projections (WMMA GEMM, 128x128 tiles)
  dim3 gqkv(HD / 128, NTOK / 128);
  gemm_f16_kernel<<<gqkv, 128, 0, stream>>>(h1_h, Wq_h, qf, NTOK, HD, HD);
  gemm_f16_kernel<<<gqkv, 128, 0, stream>>>(h1_h, Wk_h, kf, NTOK, HD, HD);
  gemm_f16_kernel<<<gqkv, 128, 0, stream>>>(h1_h, Wv_h, vf, NTOK, HD, HD);

  // 4) permute to [B,NH,S,64] f16
  permute_qkv_kernel<<<blocks(NH_), EW, 0, stream>>>(qf, qh, (int)NH_);
  permute_qkv_kernel<<<blocks(NH_), EW, 0, stream>>>(kf, kh, (int)NH_);
  permute_qkv_kernel<<<blocks(NH_), EW, 0, stream>>>(vf, vh, (int)NH_);

  // 5) causal attention (WMMA)
  dim3 ag(SEQ / 16, NHEAD, BATCH);
  attn_kernel<<<ag, 128, ATT_SMEM, stream>>>(qh, kh, vh, attn_h);

  // 6) output projection + residual
  gemm_f16_kernel<<<gqkv, 128, 0, stream>>>(attn_h, Wo_h, proj, NTOK, HD, HD);
  residual_kernel<<<blocks(NH_), EW, 0, stream>>>(x, proj, out, (int)NH_);

  // 7) LN2 + router
  ln_kernel<<<NTOK, EW, 0, stream>>>(out, ln2_g, ln2_b, h2_h);
  router_logits_kernel<<<NTOK / 8, EW, 0, stream>>>(h2_h, Wr, logits);
  routing_kernel<<<NTOK / 256, EW, 0, stream>>>(logits, we);
  lb_kernel<<<1, EW, 0, stream>>>(logits, out + NH_);

  // 8) dense MoE, expert loop
  dim3 g13(FDIM / 128, NTOK / 128);
  dim3 g2(HD / 128, NTOK / 128);
  for (int e = 0; e < NEXP; ++e) {
    const float* W1e = W1 + (size_t)e * WFH;
    const float* W3e = W3 + (size_t)e * WFH;
    const float* W2e = W2 + (size_t)e * WFH;
    cvt_f16_kernel<<<blocks(WFH), EW, 0, stream>>>(W1e, w1e_h, (int)WFH);
    cvt_f16_kernel<<<blocks(WFH), EW, 0, stream>>>(W3e, w3e_h, (int)WFH);
    cvt_f16_kernel<<<blocks(WFH), EW, 0, stream>>>(W2e, w2e_h, (int)WFH);
    gemm_f16_kernel<<<g13, 128, 0, stream>>>(h2_h, w1e_h, g1, NTOK, FDIM, HD);
    gemm_f16_kernel<<<g13, 128, 0, stream>>>(h2_h, w3e_h, g3, NTOK, FDIM, HD);
    silu_mul_kernel<<<blocks(NF_), EW, 0, stream>>>(g1, g3, hg_h, (int)NF_);
    gemm_f16_kernel<<<g2, 128, 0, stream>>>(hg_h, w2e_h, ye, NTOK, HD, FDIM);
    moe_accum_kernel<<<blocks(NH_), EW, 0, stream>>>(out, ye, we, e, (int)NH_);
  }
  (void)in_sizes; (void)n_in; (void)out_size; (void)ws_size;
}
